// MultiHeadSelfAttention_67044439491193
// MI455X (gfx1250) — matrix operand order repaired and checked by reading, not compiled, not run on hardware
//
#include <hip/hip_runtime.h>
#include <stdint.h>

// ---------------------------------------------------------------------------
// MultiHeadSelfAttention for MI455X (gfx1250, wave32, WMMA bf16 + TDM path).
// B=2, S=2048, IN_F=1024, HEADS=8, DPH=64, OUT_F=1024.
// Quirk-faithful: raw reshape => per (b,h) attention over P=4096 positions of
// 64 dims, head slices are contiguous; post-softmax /sqrt(64).
// ---------------------------------------------------------------------------

typedef __attribute__((ext_vector_type(16))) __bf16        bf16x16;
typedef __attribute__((ext_vector_type(8)))  float         f32x8;
typedef __attribute__((ext_vector_type(2)))  unsigned int  u32x2;
typedef __attribute__((ext_vector_type(4)))  unsigned int  u32x4;
typedef __attribute__((ext_vector_type(8)))  unsigned int  u32x8;

#define WMMA_BF16(A, B, C) \
  __builtin_amdgcn_wmma_f32_16x16x32_bf16(false, (A), false, (B), (short)0, (C), false, false)

__device__ __forceinline__ unsigned short f32_to_bf16(float f) {
  unsigned int u = __builtin_bit_cast(unsigned int, f);
  u += 0x7FFFu + ((u >> 16) & 1u);   // round-to-nearest-even
  return (unsigned short)(u >> 16);
}

// A fragment (16x32 bf16, M rows): lane<16 -> row=lane, K 0..7 & 16..23;
// lane>=16 -> row=lane-16, K 8..15 & 24..31.  stride in ushort elements;
// base/stride chosen so every load is 16B aligned.
__device__ __forceinline__ bf16x16 load_a_frag(const unsigned short* base, int stride) {
  const int lane = threadIdx.x & 31;
  const unsigned short* p = base + (lane & 15) * stride + (lane >> 4) * 8;
  u32x4 lo = *(const u32x4*)(p);        // K k0..k0+7
  u32x4 hi = *(const u32x4*)(p + 16);   // K k0+16..k0+23
  u32x8 w = __builtin_shufflevector(lo, hi, 0, 1, 2, 3, 4, 5, 6, 7);
  return __builtin_bit_cast(bf16x16, w);
}

// B fragment (32x16 bf16, N cols) from an N-major (transposed) tile Bt[n][k]:
// lane<16 -> N=lane, K 0..7 & 16..23; lane>=16 -> N=lane-16, K 8..15 & 24..31.
__device__ __forceinline__ bf16x16 load_b_frag(const unsigned short* base, int stride) {
  const int lane = threadIdx.x & 31;
  const unsigned short* p = base + (lane & 15) * stride + (lane >> 4) * 8;
  u32x4 lo = *(const u32x4*)(p);
  u32x4 hi = *(const u32x4*)(p + 16);
  u32x8 w = __builtin_shufflevector(lo, hi, 0, 1, 2, 3, 4, 5, 6, 7);
  return __builtin_bit_cast(bf16x16, w);
}

// ---------------------------------------------------------------------------
// Kernel 1: QKV projection.  X[4096x1024]f32 @ W[1024x1024]f32 -> bf16 out.
// Block tile 128(M) x 64(N), K-step 32, 8 waves (4x2), 16x16x32 WMMA.
// grid = (16, 32, 3); z picks Wq/Wk/Wv.
// ---------------------------------------------------------------------------
__global__ __launch_bounds__(256) void qkv_kernel(
    const float* __restrict__ X,
    const float* __restrict__ Wq, const float* __restrict__ Wk,
    const float* __restrict__ Wv,
    unsigned short* __restrict__ Qb, unsigned short* __restrict__ Kb,
    unsigned short* __restrict__ Vb) {
  __shared__ unsigned short As[128 * 48];  // row-major, stride 48 (96B, 16B-aligned)
  __shared__ unsigned short Bt[64 * 48];   // W tile transposed: Bt[n][k]

  const float* W;
  unsigned short* Out;
  if (blockIdx.z == 0)      { W = Wq; Out = Qb; }
  else if (blockIdx.z == 1) { W = Wk; Out = Kb; }
  else                      { W = Wv; Out = Vb; }

  const int m0 = blockIdx.y * 128;
  const int n0 = blockIdx.x * 64;
  const int tid  = threadIdx.x;
  const int lane = tid & 31, wave = tid >> 5;
  const int wm = wave >> 1, wn = wave & 1;       // 4x2 wave grid
  const int half = lane >> 4, nl = lane & 15;

  f32x8 zero = {0.f, 0.f, 0.f, 0.f, 0.f, 0.f, 0.f, 0.f};
  f32x8 acc[2][2] = {{zero, zero}, {zero, zero}};

  for (int kk = 0; kk < 1024; kk += 32) {
    // Stage A (fp32 -> bf16): 128x32, float4 per thread per pass.
    {
      const int r = tid >> 3;          // 0..31
      const int c = (tid & 7) * 4;     // 0..28
      for (int p = 0; p < 4; ++p) {
        const int row = p * 32 + r;
        float4 v = *(const float4*)(X + (size_t)(m0 + row) * 1024 + kk + c);
        unsigned int p0 = (unsigned)f32_to_bf16(v.x) | ((unsigned)f32_to_bf16(v.y) << 16);
        unsigned int p1 = (unsigned)f32_to_bf16(v.z) | ((unsigned)f32_to_bf16(v.w) << 16);
        u32x2 pk = {p0, p1};
        *(u32x2*)(&As[row * 48 + c]) = pk;
      }
    }
    // Stage B transposed (fp32 -> bf16): Bt[n][k] for n in [n0,n0+64), k 32.
    {
      const int k = tid >> 6;          // 0..3
      const int n = tid & 63;
      for (int p = 0; p < 8; ++p) {
        const int kr = p * 4 + k;
        Bt[n * 48 + kr] = f32_to_bf16(W[(size_t)(kk + kr) * 1024 + n0 + n]);
      }
    }
    __syncthreads();

    bf16x16 a0 = load_a_frag(&As[(wm * 32 + 0) * 48], 48);
    bf16x16 a1 = load_a_frag(&As[(wm * 32 + 16) * 48], 48);
    bf16x16 b0 = load_b_frag(&Bt[(wn * 32 + 0) * 48], 48);
    bf16x16 b1 = load_b_frag(&Bt[(wn * 32 + 16) * 48], 48);
    acc[0][0] = WMMA_BF16(a0, b0, acc[0][0]);
    acc[0][1] = WMMA_BF16(a0, b1, acc[0][1]);
    acc[1][0] = WMMA_BF16(a1, b0, acc[1][0]);
    acc[1][1] = WMMA_BF16(a1, b1, acc[1][1]);
    __syncthreads();
  }

  // Epilogue: C layout = VGPR r, lanes0-15 M=r / lanes16-31 M=r+8, N=lane&15.
  for (int tm = 0; tm < 2; ++tm)
    for (int tn = 0; tn < 2; ++tn)
      for (int r = 0; r < 8; ++r) {
        const int row = m0 + wm * 32 + tm * 16 + r + 8 * half;
        const int col = n0 + wn * 32 + tn * 16 + nl;
        Out[(size_t)row * 1024 + col] = f32_to_bf16(acc[tm][tn][r]);
      }
}

// ---------------------------------------------------------------------------
// Kernel 2: flash attention per (b,h).  Qh/Kh/Vh are contiguous [4096][64]
// bf16 slices.  Block: 128 Q rows (8 waves x 16 rows), stream 64-key chunks.
// K chunks are staged via the Tensor Data Mover (tensor_load_to_lds) with
// TDM padding reproducing the 72-element LDS row stride; tracked by TENSORcnt.
// grid = (32, 16).
// ---------------------------------------------------------------------------
__global__ __launch_bounds__(256) void attn_kernel(
    const unsigned short* __restrict__ Qb, const unsigned short* __restrict__ Kb,
    const unsigned short* __restrict__ Vb, unsigned short* __restrict__ Cb) {
  __shared__ unsigned short Ks[64 * 72];       // K chunk [k][d] (TDM dest; == B-frag layout for QK^T)
  __shared__ unsigned short Vt[64 * 72];       // V chunk transposed [d][k] for PV B-frags
  __shared__ unsigned short Pw[8 * 16 * 72];   // per-wave P tile (acc layout -> A-frag layout)

  const int bh = blockIdx.y;                   // b*8 + h
  const int b = bh >> 3, h = bh & 7;
  const size_t hb = (size_t)b * 2048 * 1024 + (size_t)h * 4096 * 64;
  const unsigned short* Qh = Qb + hb;
  const unsigned short* Kh = Kb + hb;
  const unsigned short* Vh = Vb + hb;

  const int q0 = blockIdx.x * 128;
  const int tid = threadIdx.x, lane = tid & 31, wave = tid >> 5;
  const int half = lane >> 4, nl = lane & 15;

  // Q fragments straight from global (row stride 64 bf16 = 128B, aligned).
  const unsigned short* qp = Qh + (size_t)(q0 + wave * 16) * 64;
  bf16x16 qf0 = load_a_frag(qp, 64);        // d 0..31
  bf16x16 qf1 = load_a_frag(qp + 32, 64);   // d 32..63

  f32x8 zero = {0.f, 0.f, 0.f, 0.f, 0.f, 0.f, 0.f, 0.f};
  float mrun[8], lrun[8];
  f32x8 ctx[4] = {zero, zero, zero, zero};
  for (int r = 0; r < 8; ++r) { mrun[r] = -1e30f; lrun[r] = 0.f; }

  // --- TDM descriptor (D#) for one 64x64 bf16 K tile -> Ks with row padding.
  // group0: count=1 | lds_addr | global_addr[56:0] | type=2.
  // group1: data_size=2B, pad_enable, pad_interval=4 (32 DW = one 64-elem
  //         row), pad_amount=3 (4 DW = 8 elem -> stride 72), tensor_dim0=64,
  //         tensor_dim1=4096, tile 64x64, tensor_dim0_stride=64.
  const unsigned int ks_lds = (unsigned int)(size_t)(void*)Ks;  // LDS offset = low 32 bits
  u32x4 g0 = {1u, ks_lds, 0u, 0u};
  u32x8 g1 = {(1u << 16) | (1u << 20) | (4u << 22) | (3u << 25),
              64u << 16,                 // tensor_dim0[15:0]=64
              4096u << 16,               // tensor_dim1[15:0]=4096
              64u << 16,                 // tile_dim0=64
              64u,                       // tile_dim1=64
              64u,                       // tensor_dim0_stride=64
              4096u << 16,               // tensor_dim1_stride[15:0]=4096
              0u};

  const int ldr = tid >> 2;                  // 0..63 (chunk row)
  const int ldc = (tid & 3) * 16;            // 0/16/32/48 (d offset)

  for (int kc = 0; kc < 4096; kc += 64) {
    // Kick off the TDM copy of the K tile (one instruction, one wave).
    if (wave == 0) {
      const unsigned long long ga =
          (unsigned long long)(const void*)(Kh + (size_t)kc * 64);
      g0.z = (unsigned int)ga;                                    // addr[31:0]
      g0.w = ((unsigned int)(ga >> 32) & 0x01FFFFFFu) | (2u << 30);  // addr[56:32] | type=2
      asm volatile("tensor_load_to_lds %0, %1" :: "s"(g0), "s"(g1) : "memory");
    }
    // Meanwhile all 256 threads stage V transposed: two b128 loads + scatter.
    {
      const u32x4* src = (const u32x4*)(Vh + (size_t)(kc + ldr) * 64 + ldc);
      u32x4 va = src[0], vb = src[1];
      for (int i = 0; i < 4; ++i) {
        Vt[(ldc + 2 * i + 0) * 72 + ldr] = (unsigned short)(va[i] & 0xFFFFu);
        Vt[(ldc + 2 * i + 1) * 72 + ldr] = (unsigned short)(va[i] >> 16);
        Vt[(ldc + 8 + 2 * i + 0) * 72 + ldr] = (unsigned short)(vb[i] & 0xFFFFu);
        Vt[(ldc + 8 + 2 * i + 1) * 72 + ldr] = (unsigned short)(vb[i] >> 16);
      }
    }
    if (kc + 64 < 4096)  // global_prefetch_b8 next V chunk
      __builtin_prefetch(Vh + (size_t)(kc + 64 + ldr) * 64 + ldc, 0, 0);
    if (wave == 0) __builtin_amdgcn_s_wait_tensorcnt(0);  // TENSORcnt is per-wave
    __syncthreads();

    // Scores S = Q K^T : 4 tiles of 16x16, 2 K-steps over d.
    f32x8 s[4] = {zero, zero, zero, zero};
    for (int t = 0; t < 4; ++t) {
      bf16x16 k0 = load_b_frag(&Ks[(t * 16) * 72 + 0], 72);
      bf16x16 k1 = load_b_frag(&Ks[(t * 16) * 72 + 32], 72);
      s[t] = WMMA_BF16(qf0, k0, s[t]);
      s[t] = WMMA_BF16(qf1, k1, s[t]);
    }

    // Online softmax (rows live in a fixed 16-lane half; xor masks 1..8 stay inside it).
    for (int r = 0; r < 8; ++r) {
      float mx = fmaxf(fmaxf(s[0][r], s[1][r]), fmaxf(s[2][r], s[3][r]));
      for (int off = 1; off < 16; off <<= 1) mx = fmaxf(mx, __shfl_xor(mx, off, 32));
      const float mnew  = fmaxf(mrun[r], mx);
      const float scale = __expf(mrun[r] - mnew);
      float rs = 0.f;
      for (int t = 0; t < 4; ++t) {
        const float p = __expf(s[t][r] - mnew);
        s[t][r] = p;
        rs += p;
      }
      for (int off = 1; off < 16; off <<= 1) rs += __shfl_xor(rs, off, 32);
      lrun[r] = lrun[r] * scale + rs;
      mrun[r] = mnew;
      for (int dt = 0; dt < 4; ++dt) ctx[dt][r] = ctx[dt][r] * scale;
    }

    // P: accumulator layout -> bf16 A-fragment layout via per-wave LDS tile.
    unsigned short* Pp = &Pw[wave * 16 * 72];
    for (int t = 0; t < 4; ++t)
      for (int r = 0; r < 8; ++r)
        Pp[(r + 8 * half) * 72 + t * 16 + nl] = f32_to_bf16(s[t][r]);
    // Same-wave LDS RAW: compiler inserts s_wait_dscnt; no barrier needed.

    // ctx += P @ V : 2 K-steps over key positions, 4 d-tiles.
    for (int ks = 0; ks < 2; ++ks) {
      bf16x16 pa = load_a_frag(Pp + ks * 32, 72);
      for (int dt = 0; dt < 4; ++dt) {
        bf16x16 vb = load_b_frag(&Vt[(dt * 16) * 72 + ks * 32], 72);
        ctx[dt] = WMMA_BF16(pa, vb, ctx[dt]);
      }
    }
    __syncthreads();  // protect Ks/Vt before next chunk's staging
  }

  // Finalize: ctx / (l * sqrt(64))  (post-softmax scale quirk), store bf16.
  const size_t cb = (size_t)bh * 4096 * 64;
  for (int dt = 0; dt < 4; ++dt)
    for (int r = 0; r < 8; ++r) {
      const int row = q0 + wave * 16 + r + 8 * half;
      const float v = ctx[dt][r] / (lrun[r] * 8.0f);
      Cb[cb + (size_t)row * 64 + dt * 16 + nl] = f32_to_bf16(v);
    }
}

// ---------------------------------------------------------------------------
// Kernel 3: output projection.  ctx[65536][64]bf16 @ Wo[64][1024]f32 + bo
// -> out[65536][1024]f32.  Block tile 128(M) x 64(N); K=64 (2 WMMA steps).
// grid = (16, 512).
// ---------------------------------------------------------------------------
__global__ __launch_bounds__(256) void proj_kernel(
    const unsigned short* __restrict__ Cb, const float* __restrict__ Wo,
    const float* __restrict__ bo, float* __restrict__ out) {
  __shared__ unsigned short Bt[64 * 72];  // Wo tile transposed: Bt[n][k]

  const int n0 = blockIdx.x * 64;
  const int m0 = blockIdx.y * 128;
  const int tid = threadIdx.x, lane = tid & 31, wave = tid >> 5;
  const int half = lane >> 4, nl = lane & 15;

  {
    const int k = tid >> 2;          // 0..63
    const int c = (tid & 3) * 16;    // n offset
    const float* src = Wo + (size_t)k * 1024 + n0 + c;
    for (int i = 0; i < 16; ++i) Bt[(c + i) * 72 + k] = f32_to_bf16(src[i]);
  }
  __syncthreads();

  const unsigned short* arow = Cb + (size_t)(m0 + wave * 16) * 64;
  bf16x16 a0 = load_a_frag(arow, 64);
  bf16x16 a1 = load_a_frag(arow + 32, 64);

  f32x8 zero = {0.f, 0.f, 0.f, 0.f, 0.f, 0.f, 0.f, 0.f};
  f32x8 acc[4] = {zero, zero, zero, zero};
  for (int t = 0; t < 4; ++t) {
    bf16x16 b0 = load_b_frag(&Bt[(t * 16) * 72 + 0], 72);
    bf16x16 b1 = load_b_frag(&Bt[(t * 16) * 72 + 32], 72);
    acc[t] = WMMA_BF16(a0, b0, acc[t]);
    acc[t] = WMMA_BF16(a1, b1, acc[t]);
  }

  for (int t = 0; t < 4; ++t) {
    const float bv = bo[n0 + t * 16 + nl];
    for (int r = 0; r < 8; ++r) {
      const int row = m0 + wave * 16 + r + 8 * half;
      out[(size_t)row * 1024 + n0 + t * 16 + nl] = acc[t][r] + bv;
    }
  }
}

// ---------------------------------------------------------------------------
extern "C" void kernel_launch(void* const* d_in, const int* in_sizes, int n_in,
                              void* d_out, int out_size, void* d_ws, size_t ws_size,
                              hipStream_t stream) {
  const float* X  = (const float*)d_in[0];   // [2,2048,1024]
  const float* Wq = (const float*)d_in[1];   // [1024,1024]
  const float* Wk = (const float*)d_in[2];
  const float* Wv = (const float*)d_in[3];
  const float* Wo = (const float*)d_in[4];   // [64,1024]
  const float* bo = (const float*)d_in[5];   // [1024]
  float* out = (float*)d_out;                // [2,8,4096,1024] f32

  // Workspace: Q,K,V [4096][1024] bf16 (8MB each) + ctx [16][4096][64] bf16 (8MB).
  unsigned short* Qb = (unsigned short*)d_ws;
  unsigned short* Kb = Qb + (size_t)4096 * 1024;
  unsigned short* Vb = Kb + (size_t)4096 * 1024;
  unsigned short* Cb = Vb + (size_t)4096 * 1024;

  qkv_kernel<<<dim3(16, 32, 3), 256, 0, stream>>>(X, Wq, Wk, Wv, Qb, Kb, Vb);
  attn_kernel<<<dim3(32, 16), 256, 0, stream>>>(Qb, Kb, Vb, Cb);
  proj_kernel<<<dim3(16, 512), 256, 0, stream>>>(Cb, Wo, bo, out);
}
